// FieldFormer_MHTAP_81578608820822
// MI455X (gfx1250) — compile-verified, matrix operand order checked by reading, not run on hardware
//
#include <hip/hip_runtime.h>
#include <hip/hip_bf16.h>
#include <math.h>

// ---------------- problem constants ----------------
#define NQ     2197          // number of patches (13^3)
#define DIMC   216           // input channel dim
#define KPAD   224           // DIMC padded to multiple of 32
#define KEYD   512           // H*HD
#define HHEADS 8
#define HD     64
#define HN     17576         // H*NQ
#define FX     338           // 2*13*13
#define LOUT   42
#define OUT3   (LOUT*LOUT*LOUT)      // 74088
#define ATT_OFF OUT3
#define C13_6  4826809       // 13^6
#define C13_5  371293
#define C13_4  28561
#define MTILES 138           // ceil(2197/16)
#define NTILES_PROJ 64       // 1024/16
#define INV_SQRT_DIM 0.0680413817439772f  // 1/sqrt(216)

typedef __bf16 bf16_t;
typedef bf16_t v16bf __attribute__((ext_vector_type(16)));
typedef float  v8f   __attribute__((ext_vector_type(8)));
typedef unsigned int u32x4 __attribute__((ext_vector_type(4)));
typedef int          i32x8 __attribute__((ext_vector_type(8)));
typedef int          i32x4 __attribute__((ext_vector_type(4)));

union FragU { uint4 u[2]; v16bf v; };

static __device__ __forceinline__ unsigned short f2bf(float f) {
    unsigned u = __float_as_uint(f);
    unsigned r = (u + 0x7FFFu + ((u >> 16) & 1u)) >> 16;
    return (unsigned short)r;
}

// 16-bit A/B fragment: lane<16 holds K {k0..k0+7, k0+16..k0+23},
// lane>=16 holds K {k0+8..k0+15, k0+24..k0+31} of its row.
static __device__ __forceinline__ v16bf load_frag16(const unsigned short* rowPtr,
                                                    int k0, int half) {
    FragU f;
    f.u[0] = *(const uint4*)(rowPtr + k0 + half * 8);
    f.u[1] = *(const uint4*)(rowPtr + k0 + 16 + half * 8);
    return f.v;
}

// ---------------- prep: f32 -> bf16 with zero K-padding ----------------
__global__ __launch_bounds__(256) void prep_patches_k(const float* __restrict__ p,
                                                      unsigned short* __restrict__ A) {
    int idx = blockIdx.x * 256 + threadIdx.x;
    if (idx >= NQ * KPAD) return;
    int n = idx / KPAD, c = idx - n * KPAD;
    float v = (c < DIMC) ? p[(size_t)n * DIMC + c] : 0.0f;
    A[idx] = f2bf(v);
}

__global__ __launch_bounds__(256) void prep_weights_k(const float* __restrict__ Wq,
                                                      const float* __restrict__ Wk,
                                                      unsigned short* __restrict__ W) {
    int idx = blockIdx.x * 256 + threadIdx.x;
    if (idx >= 1024 * KPAD) return;
    int o = idx / KPAD, c = idx - o * KPAD;
    float v = 0.0f;
    if (c < DIMC) v = (o < KEYD) ? Wq[(size_t)o * DIMC + c]
                                 : Wk[(size_t)(o - KEYD) * DIMC + c];
    W[idx] = f2bf(v);
}

// ---------------- projection GEMM: qk[n][o] = patches . W[o]^T  (WMMA bf16) ---
__global__ __launch_bounds__(128) void proj_wmma_k(const unsigned short* __restrict__ A,
                                                   const unsigned short* __restrict__ W,
                                                   float* __restrict__ qk) {
    int wave = threadIdx.x >> 5, lane = threadIdx.x & 31;
    int mtile = blockIdx.x;                 // 0..137
    int ntile = blockIdx.y * 4 + wave;      // 0..63
    int half = lane >> 4, r = lane & 15;
    int arow = mtile * 16 + r; if (arow > NQ - 1) arow = NQ - 1;
    const unsigned short* arp = A + (size_t)arow * KPAD;
    const unsigned short* brp = W + (size_t)(ntile * 16 + r) * KPAD;
    v8f c = {0.f,0.f,0.f,0.f,0.f,0.f,0.f,0.f};
#pragma unroll
    for (int k0 = 0; k0 < KPAD; k0 += 32) {
        v16bf a = load_frag16(arp, k0, half);
        v16bf b = load_frag16(brp, k0, half);
        c = __builtin_amdgcn_wmma_f32_16x16x32_bf16(false, a, false, b,
                                                    (short)0, c, false, false);
    }
    int col = ntile * 16 + (lane & 15);
    int rowB = mtile * 16 + 8 * half;
#pragma unroll
    for (int v = 0; v < 8; ++v) {
        int rr = rowB + v;
        if (rr < NQ) qk[(size_t)rr * 1024 + col] = c[v];
    }
}

// ---------------- per (tensor, head) batchnorm statistics ----------------
__global__ __launch_bounds__(256) void stats_k(const float* __restrict__ qk,
                                               float* __restrict__ stats) {
    int t = blockIdx.x >> 3, h = blockIdx.x & 7;
    const float* base = qk + t * KEYD + h * HD;
    float s = 0.f, s2 = 0.f;
    for (int idx = threadIdx.x; idx < NQ * HD; idx += 256) {
        int n = idx >> 6, c = idx & 63;
        float v = base[(size_t)n * 1024 + c];
        s += v; s2 += v * v;
    }
    __shared__ float red0[256], red1[256];
    red0[threadIdx.x] = s; red1[threadIdx.x] = s2;
    __syncthreads();
    for (int off = 128; off; off >>= 1) {
        if (threadIdx.x < off) {
            red0[threadIdx.x] += red0[threadIdx.x + off];
            red1[threadIdx.x] += red1[threadIdx.x + off];
        }
        __syncthreads();
    }
    if (threadIdx.x == 0) {
        const float inv = 1.0f / (float)(NQ * HD);
        float mean = red0[0] * inv;
        float var  = red1[0] * inv - mean * mean;
        stats[blockIdx.x * 2]     = mean;
        stats[blockIdx.x * 2 + 1] = rsqrtf(var + 1e-5f);
    }
}

// ------- BN apply + L2-normalize row + (1/sqrt(DIM) folded into Q) -> bf16 ----
__global__ __launch_bounds__(256) void norm_k(const float* __restrict__ qk,
                                              const float* __restrict__ stats,
                                              const float* __restrict__ gamma,
                                              const float* __restrict__ beta,
                                              unsigned short* __restrict__ QnKn) {
    int wave = threadIdx.x >> 5, lane = threadIdx.x & 31;
    int row = blockIdx.x * 8 + wave;                   // 0 .. 2*8*2197-1
    if (row >= 2 * HHEADS * NQ) return;
    int t = row / (HHEADS * NQ);
    int rem = row - t * HHEADS * NQ;
    int h = rem / NQ, n = rem - h * NQ;
    float mu = stats[(t * 8 + h) * 2], rstd = stats[(t * 8 + h) * 2 + 1];
    float g = gamma[h], b = beta[h];
    const float* src = qk + (size_t)n * 1024 + t * KEYD + h * HD;
    float y0 = (src[lane]      - mu) * rstd * g + b;
    float y1 = (src[lane + 32] - mu) * rstd * g + b;
    float s = y0 * y0 + y1 * y1;
    for (int m = 16; m; m >>= 1) s += __shfl_xor(s, m, 32);
    float inv = 1.0f / fmaxf(sqrtf(s), 1e-12f);
    if (t == 0) inv *= INV_SQRT_DIM;    // fold 1/sqrt(216) into Q
    unsigned short* dst = QnKn + ((size_t)(t * 8 + h) * NQ + n) * HD;
    dst[lane]      = f2bf(y0 * inv);
    dst[lane + 32] = f2bf(y1 * inv);
}

// ------- scores[h] = Qn . Kn^T (WMMA bf16), stored TRANSPOSED:
// ------- ST[(h*2197 + k)*2197 + q]  (q contiguous -> coalesced sparsemax) -----
__global__ __launch_bounds__(128) void scores_wmma_k(const unsigned short* __restrict__ QnKn,
                                                     float* __restrict__ ST) {
    int wave = threadIdx.x >> 5, lane = threadIdx.x & 31;
    int qtile = blockIdx.x;                  // 0..137
    int ktile = blockIdx.y * 4 + wave;       // 0..139
    int h = blockIdx.z;
    if (ktile >= MTILES) return;             // wave-uniform guard
    int half = lane >> 4, r = lane & 15;
    int qrow = qtile * 16 + r; if (qrow > NQ - 1) qrow = NQ - 1;
    int krow = ktile * 16 + r; if (krow > NQ - 1) krow = NQ - 1;
    const unsigned short* qp = QnKn + ((size_t)h * NQ + qrow) * HD;
    const unsigned short* kp = QnKn + ((size_t)(HHEADS + h) * NQ + krow) * HD;
    v8f c = {0.f,0.f,0.f,0.f,0.f,0.f,0.f,0.f};
#pragma unroll
    for (int k0 = 0; k0 < HD; k0 += 32) {
        v16bf a = load_frag16(qp, k0, half);
        v16bf b = load_frag16(kp, k0, half);
        c = __builtin_amdgcn_wmma_f32_16x16x32_bf16(false, a, false, b,
                                                    (short)0, c, false, false);
    }
    int kcol = ktile * 16 + (lane & 15);
    int qBase = qtile * 16 + 8 * half;
    if (kcol < NQ) {
        float* dst = ST + ((size_t)h * NQ + kcol) * NQ + qBase;  // 8 contiguous q
#pragma unroll
        for (int v = 0; v < 8; ++v)
            if (qBase + v < NQ) dst[v] = c[v];
    }
}

// ------- sparsemax over q: one wave per ST row (fully coalesced) --------------
#define SPX_ELEMS 69   // ceil(2197/32)
#define SPX_ITERS 36
__global__ __launch_bounds__(256) void sparsemax_k(float* __restrict__ ST) {
    int wave = threadIdx.x >> 5, lane = threadIdx.x & 31;
    int row = blockIdx.x * 8 + wave;         // 0 .. 17575
    if (row >= HN) return;
    float* base = ST + (size_t)row * NQ;
    float z[SPX_ELEMS];
    float zmax = -3.0e38f;
#pragma unroll
    for (int j = 0; j < SPX_ELEMS; ++j) {
        int q = lane + 32 * j;
        float v = (q < NQ) ? base[q] : -3.0e38f;
        z[j] = v;
        zmax = fmaxf(zmax, v);
    }
    for (int m = 16; m; m >>= 1) zmax = fmaxf(zmax, __shfl_xor(zmax, m, 32));
    // f(tau) = sum(max(z - tau, 0)) - 1 monotone decreasing; root in [zmax-1, zmax]
    float lo = zmax - 1.0f, hi = zmax;
    for (int it = 0; it < SPX_ITERS; ++it) {
        float tau = 0.5f * (lo + hi);
        float s = 0.f;
#pragma unroll
        for (int j = 0; j < SPX_ELEMS; ++j) s += fmaxf(z[j] - tau, 0.f);
        for (int m = 16; m; m >>= 1) s += __shfl_xor(s, m, 32);
        if (s >= 1.0f) lo = tau; else hi = tau;
    }
    float tau = 0.5f * (lo + hi);
#pragma unroll
    for (int j = 0; j < SPX_ELEMS; ++j) {
        int q = lane + 32 * j;
        if (q < NQ) base[q] = fmaxf(z[j] - tau, 0.f);
    }
}

// ------- transpose ST -> att via Tensor Data Mover (TDM) ----------------------
// 64x64 f32 tile DMA'd into LDS with pad (65-dword pitch -> conflict-free
// transposed reads), then coalesced store to q-major att.
__global__ __launch_bounds__(256) void transpose_tdm_k(const float* __restrict__ ST,
                                                       float* __restrict__ att) {
    __shared__ float tile[64 * 65];
    int q0 = blockIdx.x * 64;          // q tile base
    int c0 = blockIdx.y * 64;          // (h*NQ + k) tile base
    if (threadIdx.x < 32) {            // one wave issues the TDM op
        unsigned lds_off = (unsigned)(unsigned long long)(void*)&tile[0];
        unsigned long long gaddr =
            (unsigned long long)(const void*)(ST + (size_t)c0 * NQ + q0);
        int dim0 = NQ - q0;            // remaining q extent from tile start
        int dim1 = HN - c0;            // remaining row extent from tile start
        u32x4 g0 = { 1u,               // count=1, user descriptor
                     lds_off,          // lds_addr
                     (unsigned)(gaddr & 0xFFFFFFFFu),
                     (unsigned)((gaddr >> 32) & 0x1FFFFFFu) | 0x80000000u }; // type=2
        i32x8 g1 = {
            (2 << 16) | (1 << 20) | (5 << 22), // data_size=4B, pad_enable, 64dw interval, pad 1dw
            (int)((dim0 & 0xFFFF) << 16),                        // tensor_dim0[15:0]
            (int)(((unsigned)dim0 >> 16) | ((dim1 & 0xFFFF) << 16)), // dim0 hi | dim1 lo
            (int)(((unsigned)dim1 >> 16) | (64u << 16)),         // dim1 hi | tile_dim0=64
            64,                                                  // tile_dim1=64
            NQ,                                                  // tensor_dim0_stride=2197
            (int)((NQ & 0xFFFF) << 16),                          // tensor_dim1_stride lo
            0 };                                                 // stride hi
        i32x4 gz = {0, 0, 0, 0};
        i32x8 gz8 = {0, 0, 0, 0, 0, 0, 0, 0};
        __builtin_amdgcn_tensor_load_to_lds(g0, g1, gz, gz, gz8, 0);
        __builtin_amdgcn_s_wait_tensorcnt(0);
    }
    __syncthreads();
    int ci = threadIdx.x & 63;
    for (int qi = threadIdx.x >> 6; qi < 64; qi += 4) {
        int q = q0 + qi, c = c0 + ci;
        if (q < NQ && c < HN)
            att[(size_t)q * HN + c] = tile[ci * 65 + qi];
    }
}

// ------- TCL mode-1: out1[i,Y,Z] = sum_X W1[i,X] * x4[X,Y,Z] (gather-GEMM) ----
__global__ __launch_bounds__(352) void tcl1_k(const float* __restrict__ att,
                                              const float* __restrict__ W1,
                                              float* __restrict__ out1) {
    __shared__ float sW[LOUT * FX];
    __shared__ int sOff[FX];
    for (int i = threadIdx.x; i < LOUT * FX; i += 352) sW[i] = W1[i];
    for (int X = threadIdx.x; X < FX; X += 352) {
        int d0 = X / 169, rX = X - d0 * 169;
        int d3 = rX / 13, d6 = rX - d3 * 13;
        sOff[X] = d0 * (4 * C13_6) + d3 * C13_5 + d6 * 169;
    }
    __syncthreads();
    int Y = blockIdx.x;
    int d1 = Y / 169, rY = Y - d1 * 169;
    int d4 = rY / 13, d7 = rY - d4 * 13;
    int baseY = d1 * (2 * C13_6) + d4 * C13_4 + d7 * 13;
    int z = threadIdx.x;
    if (z >= FX) return;
    int d2 = z / 169, rZ = z - d2 * 169;
    int d5 = rZ / 13, d8 = rZ - d5 * 13;
    int baseYZ = baseY + d2 * C13_6 + d5 * 2197 + d8;
    float acc[LOUT];
#pragma unroll
    for (int i = 0; i < LOUT; ++i) acc[i] = 0.f;
    for (int X = 0; X < FX; ++X) {
        float v = att[(size_t)(sOff[X] + baseYZ)];
#pragma unroll
        for (int i = 0; i < LOUT; ++i) acc[i] += sW[i * FX + X] * v;
    }
#pragma unroll
    for (int i = 0; i < LOUT; ++i)
        out1[((size_t)i * FX + Y) * FX + z] = acc[i];
}

// ------- TCL mode-2: out2[i,j,Z] = sum_Y W2[j,Y] * out1[i,Y,Z] ----------------
__global__ __launch_bounds__(352) void tcl2_k(const float* __restrict__ out1,
                                              const float* __restrict__ W2,
                                              float* __restrict__ out2) {
    __shared__ float sW[LOUT * FX];
    for (int i = threadIdx.x; i < LOUT * FX; i += 352) sW[i] = W2[i];
    __syncthreads();
    int i = blockIdx.x;
    int z = threadIdx.x;
    if (z >= FX) return;
    const float* base = out1 + (size_t)i * FX * FX + z;
    float acc[LOUT];
#pragma unroll
    for (int j = 0; j < LOUT; ++j) acc[j] = 0.f;
    for (int Y = 0; Y < FX; ++Y) {
        float v = base[(size_t)Y * FX];
#pragma unroll
        for (int j = 0; j < LOUT; ++j) acc[j] += sW[j * FX + Y] * v;
    }
#pragma unroll
    for (int j = 0; j < LOUT; ++j)
        out2[((size_t)i * LOUT + j) * FX + z] = acc[j];
}

// ------- TCL mode-3 + tanh: out[i,j,k] = tanh(sum_Z W3[k,Z]*out2[i,j,Z]) ------
__global__ __launch_bounds__(64) void tcl3_k(const float* __restrict__ out2,
                                             const float* __restrict__ W3,
                                             float* __restrict__ out) {
    int i = blockIdx.x / LOUT, j = blockIdx.x - i * LOUT;
    int kk = threadIdx.x;
    if (kk >= LOUT) return;
    const float* row = out2 + (size_t)(i * LOUT + j) * FX;
    float acc = 0.f;
    for (int Z = 0; Z < FX; ++Z) acc += W3[kk * FX + Z] * row[Z];
    out[(size_t)(i * LOUT + j) * LOUT + kk] = tanhf(acc);
}

// ---------------------------------------------------------------------------
extern "C" void kernel_launch(void* const* d_in, const int* in_sizes, int n_in,
                              void* d_out, int out_size, void* d_ws, size_t ws_size,
                              hipStream_t stream) {
    const float* patches = (const float*)d_in[0];   // (1,2197,216)
    const float* Wq      = (const float*)d_in[1];   // (512,216)
    const float* Wk      = (const float*)d_in[2];   // (512,216)
    const float* gamma   = (const float*)d_in[3];   // (8,)
    const float* beta    = (const float*)d_in[4];   // (8,)
    const float* W1      = (const float*)d_in[5];   // (42,338)
    const float* W2      = (const float*)d_in[6];
    const float* W3      = (const float*)d_in[7];

    float* out = (float*)d_out;                     // tanh output (42^3)
    float* att = (float*)d_out + ATT_OFF;           // att (2197 x 17576), q-major

    // workspace carve (256B aligned); ~170 MB total (dominated by ST)
    char* ws = (char*)d_ws;
    size_t off = 0;
    auto carve = [&](size_t bytes) { char* p = ws + off; off = (off + bytes + 255) & ~(size_t)255; return p; };
    unsigned short* Abf   = (unsigned short*)carve((size_t)NQ * KPAD * 2);        // patches bf16
    unsigned short* Wcat  = (unsigned short*)carve((size_t)1024 * KPAD * 2);      // [Wq;Wk] bf16
    float*          qkbuf = (float*)carve((size_t)NQ * 1024 * 4);                 // q|k f32
    float*          stats = (float*)carve((size_t)16 * 2 * 4);                    // mean/rstd
    unsigned short* QnKn  = (unsigned short*)carve((size_t)16 * NQ * HD * 2);     // normalized bf16
    float*          ST    = (float*)carve((size_t)HN * NQ * 4);                   // scores^T, 154.5 MB
    // out1/out2 alias the ST region (ST is dead after transpose_tdm_k)
    float*          out1  = ST;
    float*          out2  = (float*)((char*)ST + (((size_t)LOUT * FX * FX * 4 + 255) & ~(size_t)255));
    (void)ws_size; (void)in_sizes; (void)n_in; (void)out_size;

    prep_patches_k<<<(NQ * KPAD + 255) / 256, 256, 0, stream>>>(patches, Abf);
    prep_weights_k<<<(1024 * KPAD + 255) / 256, 256, 0, stream>>>(Wq, Wk, Wcat);

    proj_wmma_k<<<dim3(MTILES, NTILES_PROJ / 4), 128, 0, stream>>>(Abf, Wcat, qkbuf);

    stats_k<<<16, 256, 0, stream>>>(qkbuf, stats);
    norm_k<<<(2 * HHEADS * NQ) / 8, 256, 0, stream>>>(qkbuf, stats, gamma, beta, QnKn);

    scores_wmma_k<<<dim3(MTILES, (MTILES + 3) / 4, HHEADS), 128, 0, stream>>>(QnKn, ST);

    sparsemax_k<<<HN / 8, 256, 0, stream>>>(ST);

    transpose_tdm_k<<<dim3((NQ + 63) / 64, (HN + 63) / 64), 256, 0, stream>>>(ST, att);

    tcl1_k<<<FX, 352, 0, stream>>>(att, W1, out1);
    tcl2_k<<<LOUT, 352, 0, stream>>>(out1, W2, out2);
    tcl3_k<<<LOUT * LOUT, 64, 0, stream>>>(out2, W3, out);
}
